// FlaxGPTAttention_56410100465862
// MI455X (gfx1250) — matrix operand order repaired and checked by reading, not compiled, not run on hardware
//
#include <hip/hip_runtime.h>

#define DMODEL 2048
#define NH 16
#define DH 128
#define SEQ 2048
#define NB 2

// LDS row strides (elements) chosen so 16 consecutive rows map to distinct banks
#define GSTR 40     // GEMM tiles: 128 x 32 data at stride 40 ushorts (80B, 16B-aligned, conflict-free)
#define GBUF (128 * GSTR)
#define KSTR 136    // attn K tile: 32 x 128 data at stride 136 ushorts (272B)
#define VSTR 20     // attn V tile: 128 x 16 u32 pairs at stride 20 words (80B)
#define PSTR 40     // attn P tile: 16 x 32 data at stride 40 ushorts

typedef __attribute__((ext_vector_type(16))) __bf16 v16bf;
typedef __attribute__((ext_vector_type(8)))  float v8f;
typedef __attribute__((ext_vector_type(16))) unsigned short u16x16;
typedef __attribute__((ext_vector_type(8)))  unsigned short u16x8;
typedef __attribute__((ext_vector_type(4)))  int v4i;

#if defined(__has_builtin)
#if __has_builtin(__builtin_amdgcn_global_load_async_to_lds_b128) && \
    __has_builtin(__builtin_amdgcn_s_wait_asynccnt)
#define HAVE_ASYNC_LDS 1
#endif
#endif
#ifndef HAVE_ASYNC_LDS
#define HAVE_ASYNC_LDS 0
#endif

#if HAVE_ASYNC_LDS
typedef __attribute__((address_space(1))) v4i* gas_ptr;   // global int4*
typedef __attribute__((address_space(3))) v4i* las_ptr;   // LDS int4*
#endif

template <int N>
__device__ __forceinline__ void wait_async() {
#if HAVE_ASYNC_LDS
    __builtin_amdgcn_s_wait_asynccnt(N);
#endif
}

// copy 32B global -> LDS (2 x b128). Async (ASYNCcnt) when available, else synchronous.
__device__ __forceinline__ void copy32_g2l(unsigned short* l, const unsigned short* g) {
#if HAVE_ASYNC_LDS
    __builtin_amdgcn_global_load_async_to_lds_b128((gas_ptr)g, (las_ptr)l, 0, 0);
    __builtin_amdgcn_global_load_async_to_lds_b128((gas_ptr)(g + 8), (las_ptr)(l + 8), 0, 0);
#else
    *(u16x8*)l = *(const u16x8*)g;
    *(u16x8*)(l + 8) = *(const u16x8*)(g + 8);
#endif
}

__device__ __forceinline__ unsigned short f2bf(float f) {
    unsigned int u = __float_as_uint(f);
    u += 0x7FFFu + ((u >> 16) & 1u);   // round-to-nearest-even
    return (unsigned short)(u >> 16);
}

// A-fragment: lane half kh takes 16 contiguous K (16kh..16kh+15)
__device__ __forceinline__ v16bf pack_afrag(const unsigned short* p, int kh) {
    u16x8 lo = *(const u16x8*)(p + kh * 16);
    u16x8 hi = *(const u16x8*)(p + kh * 16 + 8);
    u16x16 t;
#pragma unroll
    for (int e = 0; e < 8; ++e) { t[e] = lo[e]; t[e + 8] = hi[e]; }
    return __builtin_bit_cast(v16bf, t);
}

// B-fragment: 16 contiguous bf16 (K = 16*kh .. +15) via two aligned 16B reads
__device__ __forceinline__ v16bf load_bfrag(const unsigned short* p) {
    u16x8 lo = *(const u16x8*)p;
    u16x8 hi = *(const u16x8*)(p + 8);
    u16x16 t;
#pragma unroll
    for (int e = 0; e < 8; ++e) { t[e] = lo[e]; t[e + 8] = hi[e]; }
    return __builtin_bit_cast(v16bf, t);
}

// ---------------- elementwise fp32 -> bf16 ----------------
__global__ __launch_bounds__(256) void cvt_f32_bf16(const float* __restrict__ in,
                                                    unsigned short* __restrict__ out, int n4) {
    int i = blockIdx.x * blockDim.x + threadIdx.x;
    if (i < n4) {
        float4 v = ((const float4*)in)[i];
        ushort4 o;
        o.x = f2bf(v.x); o.y = f2bf(v.y); o.z = f2bf(v.z); o.w = f2bf(v.w);
        ((ushort4*)out)[i] = o;
    }
}

// ---------------- transpose + convert: W[k][n] fp32 -> Wt[n][k] bf16 ----------------
__global__ __launch_bounds__(256) void transpose_cvt(const float* __restrict__ in,
                                                     unsigned short* __restrict__ out,
                                                     int R, int C) {
    __shared__ float tile[32][33];
    int tx = threadIdx.x, ty = threadIdx.y;
    int c0 = blockIdx.x * 32, r0 = blockIdx.y * 32;
#pragma unroll
    for (int i = 0; i < 32; i += 8)
        tile[ty + i][tx] = in[(size_t)(r0 + ty + i) * C + (c0 + tx)];
    __syncthreads();
#pragma unroll
    for (int i = 0; i < 32; i += 8)
        out[(size_t)(c0 + ty + i) * R + (r0 + tx)] = f2bf(tile[tx][ty + i]);
}

// ---------------- shared double-buffered bf16 GEMM mainloop (128x128 block, 8 waves) ----------------
__device__ __forceinline__ void gemm_mainloop(const unsigned short* __restrict__ A,
                                              const unsigned short* __restrict__ Bm,
                                              unsigned short* As, unsigned short* Bs,
                                              int blockM, int blockN, v8f (&acc)[2][4]) {
    const int tid = threadIdx.x;
    const int wave = tid >> 5, lane = tid & 31;
    const int wm = wave & 3, wn = wave >> 2;
    const int ln16 = lane & 15, kh = lane >> 4;
    const int srow = tid >> 1, scol = (tid & 1) * 16;
    const unsigned short* gA = &A[(size_t)(blockM + srow) * DMODEL + scol];
    const unsigned short* gB = &Bm[(size_t)(blockN + srow) * DMODEL + scol];
    const int so = srow * GSTR + scol;

    // stage tile 0
    copy32_g2l(&As[so], gA);
    copy32_g2l(&Bs[so], gB);

    int ib = 0;
    for (int kb = 0; kb < DMODEL; kb += 32, ib ^= 1) {
        if (kb + 32 < DMODEL) {   // prefetch next tile into other buffer
            copy32_g2l(&As[(ib ^ 1) * GBUF + so], gA + kb + 32);
            copy32_g2l(&Bs[(ib ^ 1) * GBUF + so], gB + kb + 32);
            wait_async<4>();      // async loads retire in order: <=4 => current tile landed
        } else {
            wait_async<0>();
        }
        __syncthreads();
        const unsigned short* Ac = &As[ib * GBUF];
        const unsigned short* Bc = &Bs[ib * GBUF];
        v16bf a[2], b[4];
#pragma unroll
        for (int mi = 0; mi < 2; ++mi)
            a[mi] = pack_afrag(&Ac[(wm * 32 + mi * 16 + ln16) * GSTR], kh);
#pragma unroll
        for (int ni = 0; ni < 4; ++ni)
            b[ni] = load_bfrag(&Bc[(wn * 64 + ni * 16 + ln16) * GSTR + kh * 16]);
#pragma unroll
        for (int mi = 0; mi < 2; ++mi)
#pragma unroll
            for (int ni = 0; ni < 4; ++ni)
                acc[mi][ni] = __builtin_amdgcn_wmma_f32_16x16x32_bf16(
                    false, a[mi], false, b[ni], (short)0, acc[mi][ni], false, false);
        __syncthreads();
    }
}

// ---------------- QKV GEMM + head scatter ----------------
__global__ __launch_bounds__(256) void qkv_gemm(const unsigned short* __restrict__ Xb,
                                                const unsigned short* __restrict__ Wt,
                                                const float* __restrict__ bias,
                                                unsigned short* __restrict__ Q,
                                                unsigned short* __restrict__ K,
                                                unsigned short* __restrict__ V) {
    __shared__ __align__(32) unsigned short As[2 * GBUF];
    __shared__ __align__(32) unsigned short Bs[2 * GBUF];
    const int tid = threadIdx.x;
    const int wave = tid >> 5, lane = tid & 31;
    const int wm = wave & 3, wn = wave >> 2;
    const int blockM = blockIdx.y * 128, blockN = blockIdx.x * 128;
    const int ln16 = lane & 15, kh = lane >> 4;

    v8f acc[2][4];
#pragma unroll
    for (int i = 0; i < 2; ++i)
#pragma unroll
        for (int j = 0; j < 4; ++j) acc[i][j] = v8f{};

    gemm_mainloop(Xb, Wt, As, Bs, blockM, blockN, acc);

    // which of Q/K/V is uniform per block: 16 column-blocks each
    const int which = blockIdx.x >> 4;
    unsigned short* const dst = (which == 0) ? Q : ((which == 1) ? K : V);
    const float mul = (which == 0) ? 0.08838834764831845f : 1.0f;  // fold 1/sqrt(128) into Q
#pragma unroll
    for (int mi = 0; mi < 2; ++mi)
#pragma unroll
        for (int ni = 0; ni < 4; ++ni) {
            int n = blockN + wn * 64 + ni * 16 + ln16;
            float bn = bias[n];
            int nn = n & 2047;
            int h = nn >> 7, d = nn & 127;
#pragma unroll
            for (int r = 0; r < 8; ++r) {
                int m = blockM + wm * 32 + mi * 16 + r + kh * 8;
                int bi = m >> 11, s = m & 2047;
                dst[((size_t)(bi * NH + h) * SEQ + s) * DH + d] =
                    f2bf((acc[mi][ni][r] + bn) * mul);
            }
        }
}

// ---------------- Flash attention: block = (b,h,64 q rows), 4 waves x 16 rows ----------------
__global__ __launch_bounds__(128) void attn_kernel(const unsigned short* __restrict__ Q,
                                                   const unsigned short* __restrict__ Kq,
                                                   const unsigned short* __restrict__ V,
                                                   unsigned short* __restrict__ O) {
    __shared__ __align__(32) unsigned short Ks[32 * KSTR];
    __shared__ __align__(32) unsigned int   Vs[128 * VSTR];   // [d][keypair] 2 keys per u32
    __shared__ __align__(32) unsigned short Ps[4][16 * PSTR];

    const int tid = threadIdx.x;
    const int wave = tid >> 5, lane = tid & 31;
    const int ln16 = lane & 15, kh = lane >> 4;
    const int b = blockIdx.z, h = blockIdx.y;
    const int qblk = blockIdx.x * 64;
    const int qbase = qblk + wave * 16;

    const size_t headoff = (size_t)(b * NH + h) * SEQ * DH;
    const unsigned short* Qh = Q + headoff;
    const unsigned short* Kh = Kq + headoff;
    const unsigned short* Vh = V + headoff;

    v16bf qf[4];
    {
        const unsigned short* qp = &Qh[(size_t)(qbase + ln16) * DH];
#pragma unroll
        for (int c = 0; c < 4; ++c) qf[c] = pack_afrag(qp + c * 32, kh);
    }

    v8f o_acc[8];
#pragma unroll
    for (int i = 0; i < 8; ++i) o_acc[i] = v8f{};
    float mrow[8], lrow[8];
#pragma unroll
    for (int r = 0; r < 8; ++r) { mrow[r] = -1e30f; lrow[r] = 0.0f; }

    const int kend = qblk + 64;
    for (int kt = 0; kt < kend; kt += 32) {
        {   // stage K row-major (async path), V transposed + pair-packed (VALU)
            int row = tid >> 2, co = (tid & 3) * 32;
            const unsigned short* kg = &Kh[(size_t)(kt + row) * DH + co];
            copy32_g2l(&Ks[row * KSTR + co], kg);
            copy32_g2l(&Ks[row * KSTR + co + 16], kg + 16);
            int p = tid >> 3, vo = (tid & 7) * 16;
            u16x16 v0 = *(const u16x16*)&Vh[(size_t)(kt + 2 * p) * DH + vo];
            u16x16 v1 = *(const u16x16*)&Vh[(size_t)(kt + 2 * p + 1) * DH + vo];
#pragma unroll
            for (int e = 0; e < 16; ++e)
                Vs[(vo + e) * VSTR + p] = (unsigned int)v0[e] | ((unsigned int)v1[e] << 16);
            wait_async<0>();
        }
        __syncthreads();

        // S = Q * K^T  (2 key fragments x 4 d-chunks of 32)
        v8f sc[2];
        sc[0] = v8f{}; sc[1] = v8f{};
#pragma unroll
        for (int j = 0; j < 2; ++j)
#pragma unroll
            for (int c = 0; c < 4; ++c) {
                v16bf bf = load_bfrag(&Ks[(j * 16 + ln16) * KSTR + c * 32 + kh * 16]);
                sc[j] = __builtin_amdgcn_wmma_f32_16x16x32_bf16(
                    false, qf[c], false, bf, (short)0, sc[j], false, false);
            }

        // causal mask + online softmax (a row's 16 cols live in 16 consecutive lanes)
#pragma unroll
        for (int r = 0; r < 8; ++r) {
            int qi = qbase + r + kh * 8;
            float mx = mrow[r];
#pragma unroll
            for (int j = 0; j < 2; ++j) {
                int ki = kt + j * 16 + ln16;
                float s = (ki <= qi) ? sc[j][r] : -1e30f;
                sc[j][r] = s;
                mx = fmaxf(mx, s);
            }
#pragma unroll
            for (int off = 8; off >= 1; off >>= 1)
                mx = fmaxf(mx, __shfl_xor(mx, off, 32));
            float alpha = __expf(mrow[r] - mx);
            float rsum = 0.0f;
#pragma unroll
            for (int j = 0; j < 2; ++j) {
                float p = __expf(sc[j][r] - mx);
                sc[j][r] = p;
                rsum += p;
            }
#pragma unroll
            for (int off = 8; off >= 1; off >>= 1)
                rsum += __shfl_xor(rsum, off, 32);
            lrow[r] = lrow[r] * alpha + rsum;
            mrow[r] = mx;
#pragma unroll
            for (int i = 0; i < 8; ++i) o_acc[i][r] *= alpha;
        }

        // C-layout -> A-layout for P via wave-private LDS bounce
        unsigned short* Pw = &Ps[wave][0];
#pragma unroll
        for (int j = 0; j < 2; ++j)
#pragma unroll
            for (int r = 0; r < 8; ++r)
                Pw[(r + kh * 8) * PSTR + j * 16 + ln16] = f2bf(sc[j][r]);
        v16bf pf = pack_afrag(&Pw[ln16 * PSTR], kh);

        // O += P * V  (8 d-tiles, K=32 keys)
#pragma unroll
        for (int dt = 0; dt < 8; ++dt) {
            v16bf vf = load_bfrag((const unsigned short*)&Vs[(dt * 16 + ln16) * VSTR + kh * 8]);
            o_acc[dt] = __builtin_amdgcn_wmma_f32_16x16x32_bf16(
                false, pf, false, vf, (short)0, o_acc[dt], false, false);
        }
        __syncthreads();
    }

    // normalize + store bf16 row-major [m][DMODEL]
#pragma unroll
    for (int r = 0; r < 8; ++r) {
        float inv = 1.0f / lrow[r];
        int m = b * SEQ + qbase + r + kh * 8;
        unsigned short* orow = &O[(size_t)m * DMODEL + h * DH];
#pragma unroll
        for (int dt = 0; dt < 8; ++dt)
            orow[dt * 16 + ln16] = f2bf(o_acc[dt][r] * inv);
    }
}

// ---------------- Output projection -> fp32 + bias ----------------
__global__ __launch_bounds__(256) void out_gemm(const unsigned short* __restrict__ Ob,
                                                const unsigned short* __restrict__ Wt,
                                                const float* __restrict__ bias,
                                                float* __restrict__ out) {
    __shared__ __align__(32) unsigned short As[2 * GBUF];
    __shared__ __align__(32) unsigned short Bs[2 * GBUF];
    const int tid = threadIdx.x;
    const int wave = tid >> 5, lane = tid & 31;
    const int wm = wave & 3, wn = wave >> 2;
    const int blockM = blockIdx.y * 128, blockN = blockIdx.x * 128;
    const int ln16 = lane & 15, kh = lane >> 4;

    v8f acc[2][4];
#pragma unroll
    for (int i = 0; i < 2; ++i)
#pragma unroll
        for (int j = 0; j < 4; ++j) acc[i][j] = v8f{};

    gemm_mainloop(Ob, Wt, As, Bs, blockM, blockN, acc);

#pragma unroll
    for (int mi = 0; mi < 2; ++mi)
#pragma unroll
        for (int ni = 0; ni < 4; ++ni) {
            int n = blockN + wn * 64 + ni * 16 + ln16;
            float bn = bias[n];
#pragma unroll
            for (int r = 0; r < 8; ++r) {
                int m = blockM + wm * 32 + mi * 16 + r + kh * 8;
                out[(size_t)m * DMODEL + n] = acc[mi][ni][r] + bn;
            }
        }
}

extern "C" void kernel_launch(void* const* d_in, const int* in_sizes, int n_in,
                              void* d_out, int out_size, void* d_ws, size_t ws_size,
                              hipStream_t stream) {
    const float* x     = (const float*)d_in[0];
    const float* W_qkv = (const float*)d_in[1];
    const float* b_qkv = (const float*)d_in[2];
    const float* W_out = (const float*)d_in[3];
    const float* b_out = (const float*)d_in[4];
    float* out = (float*)d_out;

    char* ws = (char*)d_ws;
    const size_t SZ_X   = (size_t)NB * SEQ * DMODEL * 2;
    const size_t SZ_QKV = (size_t)3 * DMODEL * DMODEL * 2;
    const size_t SZ_WO  = (size_t)DMODEL * DMODEL * 2;
    unsigned short* Xb     = (unsigned short*)ws;  ws += SZ_X;
    unsigned short* Wqkv_t = (unsigned short*)ws;  ws += SZ_QKV;
    unsigned short* Wout_t = (unsigned short*)ws;  ws += SZ_WO;
    unsigned short* Qb     = (unsigned short*)ws;  ws += SZ_X;
    unsigned short* Kb     = (unsigned short*)ws;  ws += SZ_X;
    unsigned short* Vb     = (unsigned short*)ws;  ws += SZ_X;
    unsigned short* Ob     = (unsigned short*)ws;  ws += SZ_X;

    int n4 = NB * SEQ * DMODEL / 4;
    cvt_f32_bf16<<<(n4 + 255) / 256, 256, 0, stream>>>(x, Xb, n4);
    transpose_cvt<<<dim3(3 * DMODEL / 32, DMODEL / 32), dim3(32, 8), 0, stream>>>(W_qkv, Wqkv_t, DMODEL, 3 * DMODEL);
    transpose_cvt<<<dim3(DMODEL / 32, DMODEL / 32), dim3(32, 8), 0, stream>>>(W_out, Wout_t, DMODEL, DMODEL);
    qkv_gemm<<<dim3(3 * DMODEL / 128, NB * SEQ / 128), 256, 0, stream>>>(Xb, Wqkv_t, b_qkv, Qb, Kb, Vb);
    attn_kernel<<<dim3(SEQ / 64, NH, NB), 128, 0, stream>>>(Qb, Kb, Vb, Ob);
    out_gemm<<<dim3(DMODEL / 128, NB * SEQ / 128), 256, 0, stream>>>(Ob, Wout_t, b_out, out);
}
